// Net_63196148793445
// MI455X (gfx1250) — matrix operand order repaired and checked by reading, not compiled, not run on hardware
//
#include <hip/hip_runtime.h>

#define D128 128
#define GCNT 512
#define BN_EPS 1e-5f

typedef __attribute__((ext_vector_type(16))) __bf16 v16bf;
typedef __attribute__((ext_vector_type(8)))  float  v8f;

static __device__ __forceinline__ unsigned short f2bf(float f) {
    union { float f; unsigned u; } x; x.f = f;
    unsigned r = x.u + 0x7FFFu + ((x.u >> 16) & 1u);   // round-to-nearest-even
    return (unsigned short)(r >> 16);
}
static __device__ __forceinline__ unsigned pack2bf(float lo, float hi) {
    return (unsigned)f2bf(lo) | ((unsigned)f2bf(hi) << 16);
}

// ---------------------------------------------------------------------------
// Weight pre-pack: fp32 W[128x128] -> bf16 fragments in WMMA B-layout.
// Packed index: (((ct*4 + ks)*32 + lane)*16 + j)  holds  W[k, col], where
// col = ct*16 + (lane&15), k = ks*32 + (lane>>4)*16 + j.
// A lane's fragment is then 32 contiguous bytes -> two b128 loads, no VALU.
// ---------------------------------------------------------------------------
__global__ void pack_w_kernel(const float* __restrict__ Wsrc,
                              unsigned short* __restrict__ dst, long total) {
    long idx = (long)blockIdx.x * blockDim.x + threadIdx.x;
    const long st = (long)gridDim.x * blockDim.x;
    for (; idx < total; idx += st) {
        const int m    = (int)(idx >> 14);
        const int r    = (int)(idx & 16383);
        const int ct   = r >> 11;
        const int ks   = (r >> 9) & 3;
        const int lane = (r >> 4) & 31;
        const int j    = r & 15;
        const int col  = ct * 16 + (lane & 15);
        const int k    = ks * 32 + (lane >> 4) * 16 + j;
        dst[idx] = f2bf(Wsrc[(size_t)m * 16384 + (size_t)k * D128 + col]);
    }
}

// ---------------------------------------------------------------------------
// WMMA GEMM: out[M x 128] = relu(A[M x 128] @ W + bias)
// A: packed bf16 row-major. Wpk: pre-packed bf16 fragments.
// F32OUT=true -> fp32 out (feeds BN reductions); false -> bf16 out (feeds GEMM2).
// grid.x = M/16, 256 threads (8 waves); wave w owns cols [16w, 16w+16).
// ---------------------------------------------------------------------------
template <bool F32OUT>
__global__ void gemm_bf16_wmma(const unsigned short* __restrict__ Abf,
                               const unsigned short* __restrict__ Wpk,
                               const float* __restrict__ bias,
                               float* __restrict__ outf,
                               unsigned short* __restrict__ outbf) {
    const int lane    = threadIdx.x & 31;
    const int wave    = threadIdx.x >> 5;          // column tile 0..7
    const int rowBase = blockIdx.x * 16;
    const int half    = lane >> 4;
    const int lrow    = lane & 15;

    union Frag { v16bf v; uint4 q[2]; };

    // B fragments: 32 contiguous bytes per (ks, lane)
    Frag bf[4];
    #pragma unroll
    for (int ks = 0; ks < 4; ++ks) {
        const uint4* bp = (const uint4*)(Wpk + (((size_t)wave * 4 + ks) * 32 + lane) * 16);
        bf[ks].q[0] = bp[0];
        bf[ks].q[1] = bp[1];
    }

    v8f acc = {};
    const unsigned short* Arow = Abf + (size_t)(rowBase + lrow) * D128;
    #pragma unroll
    for (int ks = 0; ks < 4; ++ks) {
        // A frag (16-bit A 16x32): one contiguous 16-element K-run
        // [kb, kb+16), kb = ks*32 + half*16 (matches packed B)
        Frag af;
        const int kb = ks * 32 + half * 16;
        af.q[0] = *(const uint4*)(Arow + kb);
        af.q[1] = *(const uint4*)(Arow + kb + 8);
        acc = __builtin_amdgcn_wmma_f32_16x16x32_bf16(
            false, af.v, false, bf[ks].v, (short)0, acc, false, false);
    }

    const int col = wave * 16 + lrow;
    const float b = bias[col];
    #pragma unroll
    for (int v = 0; v < 8; ++v) {
        const int row = rowBase + half * 8 + v;
        float val = acc[v] + b;
        val = val > 0.f ? val : 0.f;
        if (F32OUT) outf [(size_t)row * D128 + col] = val;
        else        outbf[(size_t)row * D128 + col] = f2bf(val);
    }
}

// ---------------------------------------------------------------------------
// Elementwise / graph kernels
// ---------------------------------------------------------------------------
__global__ void zero_kernel(float* __restrict__ p, long n) {
    long i = (long)blockIdx.x * blockDim.x + threadIdx.x;
    const long st = (long)gridDim.x * blockDim.x;
    for (; i < n; i += st) p[i] = 0.f;
}

__global__ void copy4_kernel(float4* __restrict__ dst, const float4* __restrict__ src, long n4) {
    long i = (long)blockIdx.x * blockDim.x + threadIdx.x;
    const long st = (long)gridDim.x * blockDim.x;
    for (; i < n4; i += st) dst[i] = src[i];
}

// packed fp32 -> packed bf16 (4 elems / iter)
__global__ void f32_to_bf16_kernel(const float4* __restrict__ src,
                                   uint2* __restrict__ dst, long n4) {
    long i = (long)blockIdx.x * blockDim.x + threadIdx.x;
    const long st = (long)gridDim.x * blockDim.x;
    for (; i < n4; i += st) {
        const float4 v = src[i];
        dst[i] = make_uint2(pack2bf(v.x, v.y), pack2bf(v.z, v.w));
    }
}

// bf16(a + b) fused (feeds fc2 GEMM)
__global__ void add_bf16_kernel(const float4* __restrict__ a, const float4* __restrict__ b,
                                uint2* __restrict__ o, long n4) {
    long i = (long)blockIdx.x * blockDim.x + threadIdx.x;
    const long st = (long)gridDim.x * blockDim.x;
    for (; i < n4; i += st) {
        const float4 va = a[i], vb = b[i];
        o[i] = make_uint2(pack2bf(va.x + vb.x, va.y + vb.y),
                          pack2bf(va.z + vb.z, va.w + vb.w));
    }
}

// z[dst[e]] += h[src[e]]  (z pre-initialized with h => z = h + agg)
__global__ void edge_agg_kernel(const float* __restrict__ h,
                                const int* __restrict__ src,
                                const int* __restrict__ dst,
                                float* __restrict__ z, long nwork) {
    long i = (long)blockIdx.x * blockDim.x + threadIdx.x;
    const long st = (long)gridDim.x * blockDim.x;
    for (; i < nwork; i += st) {
        const long e = i >> 5;               // 32 chunks of 4 floats per edge
        const int  c = (int)(i & 31) * 4;
        const float4 val = *(const float4*)(h + (size_t)src[e] * D128 + c);
        float* zp = z + (size_t)dst[e] * D128 + c;
        atomicAdd(zp + 0, val.x);
        atomicAdd(zp + 1, val.y);
        atomicAdd(zp + 2, val.z);
        atomicAdd(zp + 3, val.w);
    }
}

// pool[batch[r]] += h[r]
__global__ void pool_kernel(const float* __restrict__ h, const int* __restrict__ batch,
                            float* __restrict__ pool, long nwork) {
    long i = (long)blockIdx.x * blockDim.x + threadIdx.x;
    const long st = (long)gridDim.x * blockDim.x;
    for (; i < nwork; i += st) {
        const long r = i >> 5;
        const int  c = (int)(i & 31) * 4;
        const float4 val = *(const float4*)(h + (size_t)r * D128 + c);
        float* pp = pool + (size_t)batch[r] * D128 + c;
        atomicAdd(pp + 0, val.x);
        atomicAdd(pp + 1, val.y);
        atomicAdd(pp + 2, val.z);
        atomicAdd(pp + 3, val.w);
    }
}

// per-column sum & sumsq partials (128 threads = 128 columns per block)
__global__ void colstats_kernel(const float* __restrict__ t, int M,
                                float* __restrict__ sums) {
    const int col = threadIdx.x;
    const int r0  = blockIdx.x * 256;
    int r1 = r0 + 256; if (r1 > M) r1 = M;
    float s = 0.f, s2 = 0.f;
    for (int r = r0; r < r1; ++r) {
        const float v = t[(size_t)r * D128 + col];
        s += v; s2 += v * v;
    }
    atomicAdd(&sums[col], s);
    atomicAdd(&sums[D128 + col], s2);
}

__global__ void bn_finalize_kernel(const float* __restrict__ sums,
                                   const float* __restrict__ gamma,
                                   const float* __restrict__ beta,
                                   float invM, float* __restrict__ ss) {
    const int c = threadIdx.x;
    const float mean = sums[c] * invM;
    const float var  = sums[D128 + c] * invM - mean * mean;
    const float sc   = gamma[c] * rsqrtf(var + BN_EPS);
    ss[c]        = sc;
    ss[D128 + c] = beta[c] - mean * sc;
}

// h = t*scale+shift ; pool[batch[r]] += h   (fused normalize + graph pooling)
__global__ void bn_apply_pool_kernel(const float* __restrict__ t,
                                     const float* __restrict__ ss,
                                     float* __restrict__ h,
                                     const int* __restrict__ batch,
                                     float* __restrict__ pool, long n) {
    long i = (long)blockIdx.x * blockDim.x + threadIdx.x;
    const long st = (long)gridDim.x * blockDim.x;
    for (; i < n; i += st) {
        const int  c = (int)(i & (D128 - 1));
        const long r = i >> 7;
        const float v = t[i] * ss[c] + ss[D128 + c];
        h[i] = v;
        atomicAdd(&pool[(size_t)batch[r] * D128 + c], v);
    }
}

// g = t*scale+shift ; total += g
__global__ void bn_apply_addto_kernel(const float* __restrict__ t,
                                      const float* __restrict__ ss,
                                      float* __restrict__ g,
                                      float* __restrict__ total, long n) {
    long i = (long)blockIdx.x * blockDim.x + threadIdx.x;
    const long st = (long)gridDim.x * blockDim.x;
    for (; i < n; i += st) {
        const int c = (int)(i & (D128 - 1));
        const float v = t[i] * ss[c] + ss[D128 + c];
        g[i] = v;
        total[i] += v;
    }
}

// logits = total @ linW + linb ; log_softmax. One wave per graph.
__global__ void classifier_kernel(const float* __restrict__ total,
                                  const float* __restrict__ linW,
                                  const float* __restrict__ linb,
                                  float* __restrict__ out) {
    const int gi   = blockIdx.x;
    const int lane = threadIdx.x;
    __shared__ float logits[16];
    float acc = 0.f;
    if (lane < 10) {
        const float* row = total + (size_t)gi * D128;
        for (int k = 0; k < D128; ++k) acc += row[k] * linW[k * 10 + lane];
        acc += linb[lane];
        logits[lane] = acc;
    }
    __syncthreads();
    if (lane < 10) {
        float mx = -1e30f;
        for (int j = 0; j < 10; ++j) mx = fmaxf(mx, logits[j]);
        float s = 0.f;
        for (int j = 0; j < 10; ++j) s += __expf(logits[j] - mx);
        out[gi * 10 + lane] = acc - mx - __logf(s);
    }
}

// ---------------------------------------------------------------------------
extern "C" void kernel_launch(void* const* d_in, const int* in_sizes, int n_in,
                              void* d_out, int out_size, void* d_ws, size_t ws_size,
                              hipStream_t stream) {
    const float* x      = (const float*)d_in[0];
    const int*   ei     = (const int*)d_in[1];     // [2, E]
    const int*   batch  = (const int*)d_in[2];
    const float* cW1    = (const float*)d_in[3];   // [5,128,128]
    const float* cb1    = (const float*)d_in[4];
    const float* cW2    = (const float*)d_in[5];
    const float* cb2    = (const float*)d_in[6];
    const float* cg     = (const float*)d_in[7];
    const float* cbt    = (const float*)d_in[8];
    const float* fc1W   = (const float*)d_in[9];
    const float* fc1b   = (const float*)d_in[10];
    const float* fc1g   = (const float*)d_in[11];
    const float* fc1bt  = (const float*)d_in[12];
    const float* fc2W   = (const float*)d_in[13];
    const float* fc2b   = (const float*)d_in[14];
    const float* fc2g   = (const float*)d_in[15];
    const float* fc2bt  = (const float*)d_in[16];
    const float* linW   = (const float*)d_in[17];
    const float* linb   = (const float*)d_in[18];

    const int  N  = in_sizes[0] / D128;            // 100000 (multiple of 16)
    const long E  = in_sizes[1] / 2;
    const long ND = (long)N * D128;
    const long GD = (long)GCNT * D128;

    // ---- workspace layout ----
    float* ws     = (float*)d_ws;
    float* bufA   = ws;                      // z / t2        (ND f32)
    float* bufH   = ws + ND;                 // h             (ND f32)
    unsigned short* zbf  = (unsigned short*)(ws + 2 * ND);   // ND bf16
    unsigned short* t1bf = zbf + ND;                          // ND bf16
    float* p      = (float*)(t1bf + ND);     // pool          (GD f32)
    float* g      = p + GD;
    float* total  = g + GD;
    float* smallT = total + GD;
    float* sums   = smallT + GD;             // 256
    float* ss     = sums + 2 * D128;         // 256
    unsigned short* pbf      = (unsigned short*)(ss + 2 * D128);  // GD bf16
    unsigned short* smallAbf = pbf + GD;                          // GD bf16
    unsigned short* Wpk      = smallAbf + GD;                     // 12*16384 bf16
    unsigned short* WpkC1    = Wpk;                // 5 matrices
    unsigned short* WpkC2    = Wpk + 5L * 16384;   // 5 matrices
    unsigned short* WpkF1    = Wpk + 10L * 16384;
    unsigned short* WpkF2    = Wpk + 11L * 16384;

    dim3 b256(256);

    // ---- pre-pack all weights into WMMA B-fragment bf16 layout ----
    pack_w_kernel<<<320, b256, 0, stream>>>(cW1,  WpkC1, 5L * 16384);
    pack_w_kernel<<<320, b256, 0, stream>>>(cW2,  WpkC2, 5L * 16384);
    pack_w_kernel<<<64,  b256, 0, stream>>>(fc1W, WpkF1, 16384);
    pack_w_kernel<<<64,  b256, 0, stream>>>(fc2W, WpkF2, 16384);

    // ---- g = fc1(pool(x)); total = g ----
    zero_kernel<<<64, b256, 0, stream>>>(p, GD);
    pool_kernel<<<2048, b256, 0, stream>>>(x, batch, p, (long)N * 32);
    f32_to_bf16_kernel<<<64, b256, 0, stream>>>((const float4*)p, (uint2*)pbf, GD / 4);
    gemm_bf16_wmma<true><<<GCNT / 16, b256, 0, stream>>>(pbf, WpkF1, fc1b, smallT, nullptr);
    zero_kernel<<<1, b256, 0, stream>>>(sums, 2 * D128);
    colstats_kernel<<<(GCNT + 255) / 256, 128, 0, stream>>>(smallT, GCNT, sums);
    bn_finalize_kernel<<<1, 128, 0, stream>>>(sums, fc1g, fc1bt, 1.f / GCNT, ss);
    zero_kernel<<<64, b256, 0, stream>>>(total, GD);
    bn_apply_addto_kernel<<<64, b256, 0, stream>>>(smallT, ss, g, total, GD);

    const float* h = x;
    for (int i = 0; i < 5; ++i) {
        // z = h + sum_{edges} h[src] -> dst  (fp32 atomics), then bf16 pack
        copy4_kernel<<<4096, b256, 0, stream>>>((float4*)bufA, (const float4*)h, ND / 4);
        edge_agg_kernel<<<8192, b256, 0, stream>>>(h, ei, ei + E, bufA, E * 32);
        f32_to_bf16_kernel<<<4096, b256, 0, stream>>>((const float4*)bufA, (uint2*)zbf, ND / 4);

        // MLP: relu(z@W1+b1) [bf16 out] -> relu(.@W2+b2) [f32 out] -> BN -> h
        gemm_bf16_wmma<false><<<N / 16, b256, 0, stream>>>(zbf, WpkC1 + (size_t)i * 16384,
                                                           cb1 + i * D128, nullptr, t1bf);
        gemm_bf16_wmma<true><<<N / 16, b256, 0, stream>>>(t1bf, WpkC2 + (size_t)i * 16384,
                                                          cb2 + i * D128, bufA, nullptr);
        zero_kernel<<<1, b256, 0, stream>>>(sums, 2 * D128);
        colstats_kernel<<<(N + 255) / 256, 128, 0, stream>>>(bufA, N, sums);
        bn_finalize_kernel<<<1, 128, 0, stream>>>(sums, cg + i * D128, cbt + i * D128,
                                                  1.f / (float)N, ss);
        zero_kernel<<<64, b256, 0, stream>>>(p, GD);
        bn_apply_pool_kernel<<<8192, b256, 0, stream>>>(bufA, ss, bufH, batch, p, ND);
        h = bufH;

        // g = fc2(g + pool(h)); total += g
        add_bf16_kernel<<<64, b256, 0, stream>>>((const float4*)g, (const float4*)p,
                                                 (uint2*)smallAbf, GD / 4);
        gemm_bf16_wmma<true><<<GCNT / 16, b256, 0, stream>>>(smallAbf, WpkF2, fc2b, smallT, nullptr);
        zero_kernel<<<1, b256, 0, stream>>>(sums, 2 * D128);
        colstats_kernel<<<(GCNT + 255) / 256, 128, 0, stream>>>(smallT, GCNT, sums);
        bn_finalize_kernel<<<1, 128, 0, stream>>>(sums, fc2g, fc2bt, 1.f / GCNT, ss);
        bn_apply_addto_kernel<<<64, b256, 0, stream>>>(smallT, ss, g, total, GD);
    }

    classifier_kernel<<<GCNT, 32, 0, stream>>>(total, linW, linb, (float*)d_out);
}
